// Block_62156766708387
// MI455X (gfx1250) — matrix operand order repaired and checked by reading, not compiled, not run on hardware
//
#include <hip/hip_runtime.h>
#include <hip/hip_bf16.h>
#include <math.h>

// ---------------------------------------------------------------------------
// Problem constants (match reference)
// ---------------------------------------------------------------------------
#define BATCH   8
#define SEQ     1024
#define DMODEL  1024
#define NHEAD   16
#define HD      64          // head dim
#define FFN     4096
#define NEXP    8
#define NTOK    (BATCH*SEQ) // 8192
#define LN_EPS  1e-5f

typedef __attribute__((ext_vector_type(16))) __bf16 v16bf;
typedef __attribute__((ext_vector_type(8)))  float  v8f;

#define WMMA_BF16(a,b,c) \
  __builtin_amdgcn_wmma_f32_16x16x32_bf16(false,(a),false,(b),(short)0,(c),false,false)

// fp32 -> bf16 round-to-nearest-even
__device__ __forceinline__ __bf16 f2bf(float f) {
  unsigned u = __builtin_bit_cast(unsigned, f);
  unsigned r = u + 0x7FFFu + ((u >> 16) & 1u);
  unsigned short h = (unsigned short)(r >> 16);
  return __builtin_bit_cast(__bf16, h);
}

__device__ __forceinline__ float wave_max(float v) {
  #pragma unroll
  for (int o = 16; o > 0; o >>= 1) v = fmaxf(v, __shfl_xor(v, o, 32));
  return v;
}
__device__ __forceinline__ float wave_sum(float v) {
  #pragma unroll
  for (int o = 16; o > 0; o >>= 1) v += __shfl_xor(v, o, 32);
  return v;
}

// ---------------------------------------------------------------------------
// LayerNorm: one 256-thread block per row of D=1024
// ---------------------------------------------------------------------------
__global__ __launch_bounds__(256)
void ln_kernel(const float* __restrict__ x,
               const float* __restrict__ g,
               const float* __restrict__ b,
               float* __restrict__ out) {
  int row = blockIdx.x;
  const float* xr = x + (size_t)row * DMODEL;
  float* orow = out + (size_t)row * DMODEL;
  __shared__ float red[10];
  int t = threadIdx.x, wid = t >> 5, lane = t & 31;

  float v4[4];
  float s = 0.f;
  #pragma unroll
  for (int i = 0; i < 4; ++i) { v4[i] = xr[t + i*256]; s += v4[i]; }
  s = wave_sum(s);
  if (lane == 0) red[wid] = s;
  __syncthreads();
  if (t == 0) { float m = 0.f; for (int i = 0; i < 8; ++i) m += red[i]; red[8] = m * (1.f/DMODEL); }
  __syncthreads();
  float mu = red[8];

  float vs = 0.f;
  #pragma unroll
  for (int i = 0; i < 4; ++i) { float d = v4[i] - mu; vs += d*d; }
  vs = wave_sum(vs);
  if (lane == 0) red[wid] = vs;
  __syncthreads();
  if (t == 0) { float m = 0.f; for (int i = 0; i < 8; ++i) m += red[i]; red[9] = m * (1.f/DMODEL); }
  __syncthreads();
  float rstd = rsqrtf(red[9] + LN_EPS);

  #pragma unroll
  for (int i = 0; i < 4; ++i) {
    int c = t + i*256;
    orow[c] = (v4[i] - mu) * rstd * g[c] + b[c];
  }
}

// ---------------------------------------------------------------------------
// bf16-WMMA GEMM, compile-time specialized:
//   MODE 0: C = A@B + bias                       (full tiles, M % 128 == 0)
//   MODE 1: C = A@B + bias + residual            (full tiles)
//   MODE 2: C = gelu(A[glist]@B + bias), M=*cnt  (gathered rows via LDS idx)
//   MODE 3: C[glist[r]] += scales[r]*(A@B+bias), M=*cnt (A positional)
// Block tile 128x64, 256 threads = 8 waves (4x2), wave tile 32x32.
// ---------------------------------------------------------------------------
#define BM 128
#define BN 64
#define BK 32
#define AST 34
#define BST 68

template <int MODE>
__global__ __launch_bounds__(256)
void gemm_kernel(const float* __restrict__ A,
                 const float* __restrict__ Bw,
                 const float* __restrict__ bias,
                 const float* __restrict__ residual,
                 float* __restrict__ C,
                 int M, int Nc, int Kc,
                 const int*   __restrict__ cnt,
                 const int*   __restrict__ glist,
                 const float* __restrict__ scales) {
  if constexpr (MODE >= 2) M = *cnt;
  int m0 = blockIdx.x * BM;
  if (m0 >= M) return;
  int n0 = blockIdx.y * BN;

  __shared__ __bf16 As[BM * AST];
  __shared__ __bf16 Bs[BK * BST];
  __shared__ int    gidx[BM];

  int t = threadIdx.x;
  int wid = t >> 5, lane = t & 31;
  int wm = wid >> 1, wn = wid & 1;        // 4x2 wave grid
  int lm = lane & 15, half = lane >> 4;

  if constexpr (MODE >= 2) {
    if (t < BM) {
      int row = m0 + t;
      gidx[t] = (row < M) ? glist[row] : 0;
    }
    __syncthreads();
  }

  v8f acc[2][2] = {};

  for (int k0 = 0; k0 < Kc; k0 += BK) {
    // stage A tile (fp32 -> bf16); addresses branch-free
    #pragma unroll
    for (int i = 0; i < 16; ++i) {
      int idx = i * 256 + t;
      int r = idx >> 5, c = idx & 31;
      int arow;
      if constexpr (MODE == 2) arow = gidx[r];
      else                     arow = m0 + r;    // always in-bounds rows
      As[r * AST + c] = f2bf(A[(size_t)arow * Kc + k0 + c]);
    }
    // stage B tile
    #pragma unroll
    for (int i = 0; i < 8; ++i) {
      int idx = i * 256 + t;
      int r = idx >> 6, c = idx & 63;
      Bs[r * BST + c] = f2bf(Bw[(size_t)(k0 + r) * Nc + n0 + c]);
    }
    if (k0 + BK < Kc) {  // warm L2 for next B tile (global_prefetch_b8)
      __builtin_prefetch(Bw + (size_t)(k0 + BK + (t >> 3)) * Nc + n0 + (t & 7) * 8, 0, 1);
    }
    __syncthreads();

    v16bf af[2], bfr[2];
    #pragma unroll
    for (int i = 0; i < 2; ++i) {
      int row = wm * 32 + i * 16 + lm;
      #pragma unroll
      for (int e = 0; e < 16; ++e) {
        int k = 16 * half + e;
        af[i][e] = As[row * AST + k];
      }
    }
    #pragma unroll
    for (int j = 0; j < 2; ++j) {
      int col = wn * 32 + j * 16 + lm;
      #pragma unroll
      for (int e = 0; e < 16; ++e) {
        int k = 16 * half + e;
        bfr[j][e] = Bs[k * BST + col];
      }
    }
    #pragma unroll
    for (int i = 0; i < 2; ++i)
      #pragma unroll
      for (int j = 0; j < 2; ++j)
        acc[i][j] = WMMA_BF16(af[i], bfr[j], acc[i][j]);
    __syncthreads();
  }

  // epilogue
  #pragma unroll
  for (int i = 0; i < 2; ++i) {
    #pragma unroll
    for (int j = 0; j < 2; ++j) {
      int col = n0 + wn * 32 + j * 16 + lm;
      #pragma unroll
      for (int r = 0; r < 8; ++r) {
        int rl  = wm * 32 + i * 16 + r + 8 * half;  // local row 0..127
        int row = m0 + rl;
        float v = acc[i][j][r] + bias[col];
        if constexpr (MODE == 0) {
          C[(size_t)row * Nc + col] = v;
        } else if constexpr (MODE == 1) {
          C[(size_t)row * Nc + col] = v + residual[(size_t)row * Nc + col];
        } else if constexpr (MODE == 2) {
          if (row < M) {
            v = 0.5f * v * (1.f + erff(v * 0.70710678118654752f));
            C[(size_t)row * Nc + col] = v;
          }
        } else {  // MODE == 3
          if (row < M) {
            int orow = gidx[rl];
            C[(size_t)orow * Nc + col] += scales[row] * v;
          }
        }
      }
    }
  }
}

// ---------------------------------------------------------------------------
// Attention: one block per (64 queries, head, batch). Full 64x1024 fp32 score
// tile resident in LDS (CDNA5: 320KB/WGP). WMMA for QK^T and PV.
// ---------------------------------------------------------------------------
#define SSTR 1032   // padded row stride (floats) -> conflict-free A-frag loads

__global__ __launch_bounds__(256)
void attn_kernel(const float* __restrict__ Q,
                 const float* __restrict__ Km,
                 const float* __restrict__ Vm,
                 float* __restrict__ Y) {
  int q0 = blockIdx.x * 64;
  int h  = blockIdx.y;
  int bb = blockIdx.z;

  extern __shared__ char smem[];
  float*  S  = (float*)smem;                 // 64 * SSTR fp32
  __bf16* Qs = (__bf16*)(S + 64 * SSTR);     // 64 x 68
  __bf16* Ks = Qs + 64 * 68;                 // 32 x 68
  __bf16* Vs = Ks + 32 * 68;                 // 32 x 68

  int t = threadIdx.x, wid = t >> 5, lane = t & 31;
  int lm = lane & 15, half = lane >> 4;
  size_t base = ((size_t)bb * SEQ) * DMODEL + (size_t)h * HD;
  int kmax = q0 + 64;   // causal upper bound for this block

  // stage Q tile
  #pragma unroll
  for (int i = 0; i < 16; ++i) {
    int idx = i * 256 + t;
    int r = idx >> 6, c = idx & 63;
    Qs[r * 68 + c] = f2bf(Q[base + (size_t)(q0 + r) * DMODEL + c]);
  }
  __syncthreads();

  // ---- S = (Q K^T) * 1/sqrt(hd) ----
  {
    int qsub = wid & 3, ksub = wid >> 2;  // 4 q-subtiles x 2 key-subtiles
    v16bf a0, a1;
    #pragma unroll
    for (int e = 0; e < 16; ++e) {
      int k = 16 * half + e;
      a0[e] = Qs[(qsub * 16 + lm) * 68 + k];
      a1[e] = Qs[(qsub * 16 + lm) * 68 + 32 + k];
    }
    for (int k0 = 0; k0 < kmax; k0 += 32) {
      #pragma unroll
      for (int i = 0; i < 8; ++i) {
        int idx = i * 256 + t;
        int r = idx >> 6, c = idx & 63;
        Ks[r * 68 + c] = f2bf(Km[base + (size_t)(k0 + r) * DMODEL + c]);
      }
      __syncthreads();
      v16bf b0, b1;
      #pragma unroll
      for (int e = 0; e < 16; ++e) {
        int kk = 16 * half + e;
        b0[e] = Ks[(ksub * 16 + lm) * 68 + kk];        // d = 0..31
        b1[e] = Ks[(ksub * 16 + lm) * 68 + 32 + kk];   // d = 32..63
      }
      v8f c = {};
      c = WMMA_BF16(a0, b0, c);
      c = WMMA_BF16(a1, b1, c);
      #pragma unroll
      for (int r = 0; r < 8; ++r) {
        int m = r + 8 * half;
        S[(qsub * 16 + m) * SSTR + k0 + ksub * 16 + lm] = c[r] * 0.125f;
      }
      __syncthreads();
    }
  }
  __syncthreads();

  // ---- row softmax (causal: cols <= q valid; zero-fill rest of kmax) ----
  for (int rr = 0; rr < 8; ++rr) {
    int r = wid * 8 + rr;
    int q = q0 + r;
    float* srow = S + r * SSTR;
    float mx = -1e30f;
    for (int j = lane; j <= q; j += 32) mx = fmaxf(mx, srow[j]);
    mx = wave_max(mx);
    float sum = 0.f;
    for (int j = lane; j <= q; j += 32) { float p = __expf(srow[j] - mx); srow[j] = p; sum += p; }
    sum = wave_sum(sum);
    float inv = 1.f / sum;
    for (int j = lane; j < kmax; j += 32) srow[j] = (j <= q) ? srow[j] * inv : 0.f;
  }
  __syncthreads();

  // ---- Y = P V ----
  {
    int qsub = wid & 3, dpair = wid >> 2;  // each wave: 16q x 32 output cols
    v8f c0 = {}, c1 = {};
    for (int k0 = 0; k0 < kmax; k0 += 32) {
      #pragma unroll
      for (int i = 0; i < 8; ++i) {
        int idx = i * 256 + t;
        int r = idx >> 6, cc = idx & 63;
        Vs[r * 68 + cc] = f2bf(Vm[base + (size_t)(k0 + r) * DMODEL + cc]);
      }
      __syncthreads();
      v16bf a, b0, b1;
      #pragma unroll
      for (int e = 0; e < 16; ++e) {
        int k = 16 * half + e;
        a[e] = f2bf(S[(qsub * 16 + lm) * SSTR + k0 + k]);
      }
      #pragma unroll
      for (int e = 0; e < 16; ++e) {
        int kk = 16 * half + e;
        b0[e] = Vs[kk * 68 + (dpair * 2 + 0) * 16 + lm];
        b1[e] = Vs[kk * 68 + (dpair * 2 + 1) * 16 + lm];
      }
      c0 = WMMA_BF16(a, b0, c0);
      c1 = WMMA_BF16(a, b1, c1);
      __syncthreads();
    }
    #pragma unroll
    for (int r = 0; r < 8; ++r) {
      int m = r + 8 * half;
      size_t orow = base + (size_t)(q0 + qsub * 16 + m) * DMODEL;
      Y[orow + (dpair * 2 + 0) * 16 + lm] = c0[r];
      Y[orow + (dpair * 2 + 1) * 16 + lm] = c1[r];
    }
  }
}

// ---------------------------------------------------------------------------
// Gate: one wave per token. Logits, top-2, softmax, append to expert lists.
// ---------------------------------------------------------------------------
__global__ __launch_bounds__(256)
void gate_kernel(const float* __restrict__ h2,
                 const float* __restrict__ gW,
                 const float* __restrict__ gb,
                 int* __restrict__ counts,
                 int* __restrict__ lists,
                 float* __restrict__ wgts) {
  int tok = (blockIdx.x * blockDim.x + threadIdx.x) >> 5;
  int lane = threadIdx.x & 31;
  if (tok >= NTOK) return;
  const float* xr = h2 + (size_t)tok * DMODEL;
  float acc[NEXP] = {};
  for (int d = lane; d < DMODEL; d += 32) {
    float xv = xr[d];
    const float* wr = gW + (size_t)d * NEXP;
    #pragma unroll
    for (int e = 0; e < NEXP; ++e) acc[e] += xv * wr[e];
  }
  #pragma unroll
  for (int e = 0; e < NEXP; ++e) acc[e] = wave_sum(acc[e]);
  if (lane == 0) {
    int i0 = -1, i1 = -1;
    float v0 = -1e30f, v1 = -1e30f;
    #pragma unroll
    for (int e = 0; e < NEXP; ++e) {
      float v = acc[e] + gb[e];
      if (v > v0) { v1 = v0; i1 = i0; v0 = v; i0 = e; }
      else if (v > v1) { v1 = v; i1 = e; }
    }
    float e1 = __expf(v1 - v0);
    float inv = 1.f / (1.f + e1);
    int p0 = atomicAdd(&counts[i0], 1);
    lists[i0 * NTOK + p0] = tok;  wgts[i0 * NTOK + p0] = inv;
    int p1 = atomicAdd(&counts[i1], 1);
    lists[i1 * NTOK + p1] = tok;  wgts[i1 * NTOK + p1] = e1 * inv;
  }
}

__global__ void zero_counts_kernel(int* __restrict__ c) {
  if (threadIdx.x < NEXP) c[threadIdx.x] = 0;
}

// ---------------------------------------------------------------------------
// Host launch sequence
// ---------------------------------------------------------------------------
extern "C" void kernel_launch(void* const* d_in, const int* in_sizes, int n_in,
                              void* d_out, int out_size, void* d_ws, size_t ws_size,
                              hipStream_t stream) {
  const float* x     = (const float*)d_in[0];
  const float* ln1_g = (const float*)d_in[1];
  const float* ln1_b = (const float*)d_in[2];
  const float* ln2_g = (const float*)d_in[3];
  const float* ln2_b = (const float*)d_in[4];
  const float* Wq = (const float*)d_in[5];   const float* bq = (const float*)d_in[6];
  const float* Wk = (const float*)d_in[7];   const float* bk = (const float*)d_in[8];
  const float* Wv = (const float*)d_in[9];   const float* bv = (const float*)d_in[10];
  const float* Wp = (const float*)d_in[11];  const float* bp = (const float*)d_in[12];
  const float* gW = (const float*)d_in[13];  const float* gb = (const float*)d_in[14];
  const float* eW1 = (const float*)d_in[15]; const float* eb1 = (const float*)d_in[16];
  const float* eW2 = (const float*)d_in[17]; const float* eb2 = (const float*)d_in[18];
  float* out = (float*)d_out;

  const size_t ND = (size_t)NTOK * DMODEL;
  float* ws   = (float*)d_ws;
  float* h1   = ws;
  float* Qb   = h1 + ND;
  float* Kb   = Qb + ND;
  float* Vb   = Kb + ND;
  float* Yb   = Vb + ND;
  float* h2   = Yb + ND;
  float* mid  = h2 + ND;                               // NTOK * FFN
  float* wgts = mid + (size_t)NTOK * FFN;              // NEXP * NTOK
  int*   lists  = (int*)(wgts + (size_t)NEXP * NTOK);  // NEXP * NTOK
  int*   counts = lists + (size_t)NEXP * NTOK;         // NEXP

  dim3 blk(256);
  dim3 gQKV(NTOK / BM, DMODEL / BN);   // 64 x 16
  dim3 gFFN1(NTOK / BM, FFN / BN);     // 64 x 64
  dim3 gFFN2(NTOK / BM, DMODEL / BN);  // 64 x 16

  // x1 = x + attn(ln1(x))
  ln_kernel<<<NTOK, blk, 0, stream>>>(x, ln1_g, ln1_b, h1);
  gemm_kernel<0><<<gQKV, blk, 0, stream>>>(h1, Wq, bq, nullptr, Qb, NTOK, DMODEL, DMODEL,
                                           nullptr, nullptr, nullptr);
  gemm_kernel<0><<<gQKV, blk, 0, stream>>>(h1, Wk, bk, nullptr, Kb, NTOK, DMODEL, DMODEL,
                                           nullptr, nullptr, nullptr);
  gemm_kernel<0><<<gQKV, blk, 0, stream>>>(h1, Wv, bv, nullptr, Vb, NTOK, DMODEL, DMODEL,
                                           nullptr, nullptr, nullptr);

  size_t attn_lds = (size_t)64 * SSTR * sizeof(float) +
                    (size_t)(64 * 68 + 2 * 32 * 68) * sizeof(__bf16);
  attn_kernel<<<dim3(SEQ / 64, NHEAD, BATCH), blk, attn_lds, stream>>>(Qb, Kb, Vb, Yb);

  gemm_kernel<1><<<gQKV, blk, 0, stream>>>(Yb, Wp, bp, x, out, NTOK, DMODEL, DMODEL,
                                           nullptr, nullptr, nullptr);

  // x2 = x1 + moe(ln2(x1)), routed top-2
  ln_kernel<<<NTOK, blk, 0, stream>>>(out, ln2_g, ln2_b, h2);
  zero_counts_kernel<<<1, 32, 0, stream>>>(counts);
  gate_kernel<<<NTOK / 8, blk, 0, stream>>>(h2, gW, gb, counts, lists, wgts);

  for (int e = 0; e < NEXP; ++e) {
    const float* W1 = eW1 + (size_t)e * DMODEL * FFN;
    const float* b1 = eb1 + (size_t)e * FFN;
    const float* W2 = eW2 + (size_t)e * FFN * DMODEL;
    const float* b2 = eb2 + (size_t)e * DMODEL;
    const int*   le = lists + (size_t)e * NTOK;
    const float* we = wgts + (size_t)e * NTOK;
    // mid = gelu(h2[gather] @ W1 + b1), M = counts[e]
    gemm_kernel<2><<<gFFN1, blk, 0, stream>>>(h2, W1, b1, nullptr, mid, NTOK, FFN, DMODEL,
                                              counts + e, le, nullptr);
    // out[scatter] += w * (mid @ W2 + b2), M = counts[e]
    gemm_kernel<3><<<gFFN2, blk, 0, stream>>>(mid, W2, b2, nullptr, out, NTOK, DMODEL, FFN,
                                              counts + e, le, we);
  }
}
